// EMA_35390530519891
// MI455X (gfx1250) — matrix operand order repaired and checked by reading, not compiled, not run on hardware
//
#include <hip/hip_runtime.h>

typedef __attribute__((ext_vector_type(16))) __bf16 v16bf;
typedef __attribute__((ext_vector_type(8)))  float  v8f;

#define BATCH 64
#define SEQ_T 512
#define DDIM  512
#define ROWS  (BATCH * SEQ_T)   // 32768
#define SCAN_BLOCKS 16          // 16 blocks x 8 waves = 128 waves = one 16x16 tile pair each

// ---------------- fast activation helpers (TRANS ops: v_exp, v_rcp) ----------
__device__ __forceinline__ float fast_exp(float x) {
  return __builtin_amdgcn_exp2f(x * 1.4426950408889634f);
}
__device__ __forceinline__ float sigmoid_f(float x) {
  return __builtin_amdgcn_rcpf(1.0f + fast_exp(-x));
}
__device__ __forceinline__ float tanh_f(float x) {
  return 2.0f * sigmoid_f(2.0f * x) - 1.0f;
}

// ---------------- WMMA fragment loaders (16x16x32 bf16, contiguous k order) --------
// A: 16x32 tile, row-major source with leading dim ld.
// lane L: M = L%16, half = L/16. VGPR v holds K = 16*half + 2v + {0,1}.
__device__ __forceinline__ v16bf load_frag_a(const __bf16* tile, int ld) {
  int lane = threadIdx.x & 31;
  int m = lane & 15, half = lane >> 4;
  const __bf16* p = tile + m * ld + half * 16;
  v16bf f;
#pragma unroll
  for (int v = 0; v < 8; ++v) {
    int kb = 2 * v;
    f[2 * v]     = p[kb];
    f[2 * v + 1] = p[kb + 1];
  }
  return f;
}

// B: 32x16 tile where B[k][n] = W[n][k], W row-major [N][K] with leading dim ld.
// lane L: N = L%16, half = L/16; VGPR v holds K = 16*half + 2v + {0,1} -> 16 contiguous bf16/lane.
__device__ __forceinline__ v16bf load_frag_b(const __bf16* wt, int ld) {
  int lane = threadIdx.x & 31;
  int n = lane & 15, half = lane >> 4;
  const __bf16* p = wt + n * ld + half * 16;
  v16bf f;
#pragma unroll
  for (int e = 0; e < 16; ++e) f[e] = p[e];
  return f;
}

__device__ __forceinline__ v8f wmma_bf16(v16bf a, v16bf b, v8f c) {
  return __builtin_amdgcn_wmma_f32_16x16x32_bf16(false, a, false, b, (short)0, c, false, false);
}

// ---------------- conversion kernels -----------------------------------------
__global__ __launch_bounds__(256) void ema_cvt_kernel(const float* __restrict__ src,
                                                      __bf16* __restrict__ dst, int n) {
  int i = blockIdx.x * 256 + threadIdx.x;
  if (i < n) dst[i] = (__bf16)src[i];
}

// dst[r*cols + c] = (bf16) src[r*src_ld + col_off + c]
__global__ __launch_bounds__(256) void ema_cvt_slice_kernel(const float* __restrict__ src,
                                                            __bf16* __restrict__ dst,
                                                            int rows, int cols, int src_ld, int coff) {
  int i = blockIdx.x * 256 + threadIdx.x;
  if (i >= rows * cols) return;
  int r = i / cols, c = i - r * cols;
  dst[i] = (__bf16)src[r * src_ld + coff + c];
}

// ---------------- phase 1: Xa = b @ Wa2^T + ba ; Xd = b @ Wd2^T + bd ---------
// 2 m-tile register blocking: A pair reused across both weight matrices.
__global__ __launch_bounds__(256) void ema_pre_kernel(
    const __bf16* __restrict__ bb, const __bf16* __restrict__ Wa2,
    const __bf16* __restrict__ Wd2, const float* __restrict__ ba,
    const float* __restrict__ bd, float* __restrict__ Xa, float* __restrict__ Xd) {
  int wid = blockIdx.x * 8 + (threadIdx.x >> 5);
  int mtp = wid >> 5;  // 0..1023 (32768/32 row-pair tiles)
  int nt  = wid & 31;  // 0..31
  int lane = threadIdx.x & 31;
  int n = lane & 15, half = lane >> 4;

  const __bf16* a0 = bb + (size_t)(mtp * 32) * DDIM;
  const __bf16* a1 = a0 + (size_t)16 * DDIM;
  const __bf16* wa = Wa2 + (size_t)(nt * 16) * DDIM;
  const __bf16* wd = Wd2 + (size_t)(nt * 16) * DDIM;
  v8f aA0 = {}, aA1 = {}, aD0 = {}, aD1 = {};
  for (int k = 0; k < DDIM; k += 32) {
    v16bf af0 = load_frag_a(a0 + k, DDIM);
    v16bf af1 = load_frag_a(a1 + k, DDIM);
    v16bf f1 = load_frag_b(wa + k, DDIM);
    v16bf f2 = load_frag_b(wd + k, DDIM);
    aA0 = wmma_bf16(af0, f1, aA0);
    aA1 = wmma_bf16(af1, f1, aA1);
    aD0 = wmma_bf16(af0, f2, aD0);
    aD1 = wmma_bf16(af1, f2, aD1);
  }
  int col = nt * 16 + n;
  float bav = ba[col], bdv = bd[col];
#pragma unroll
  for (int i = 0; i < 8; ++i) {
    int row0 = mtp * 32 + i + half * 8;  // C layout: M = vgpr + 8*(lane>=16)
    Xa[row0 * DDIM + col] = aA0[i] + bav;
    Xd[row0 * DDIM + col] = aD0[i] + bdv;
    Xa[(row0 + 16) * DDIM + col] = aA1[i] + bav;
    Xd[(row0 + 16) * DDIM + col] = aD1[i] + bdv;
  }
}

// ---------------- device-scope barrier for persistent scan -------------------
__device__ __forceinline__ void gsync(unsigned int* bar, unsigned int target) {
  __syncthreads();
  if (threadIdx.x == 0) {
    __hip_atomic_fetch_add(bar, 1u, __ATOMIC_RELEASE, __HIP_MEMORY_SCOPE_AGENT);
    while (__hip_atomic_load(bar, __ATOMIC_ACQUIRE, __HIP_MEMORY_SCOPE_AGENT) < target)
      __builtin_amdgcn_s_sleep(2);
  }
  __syncthreads();
}

// ---------------- phase 2: sequential scan (persistent, 16 blocks) -----------
// Weights for this block's 8 n-tiles (Wa1+Wd1 = 256 KiB) staged once in the
// 320 KiB WGP LDS; B-fragments then come from ds_load on the critical path.
__global__ __launch_bounds__(256) void ema_scan_kernel(
    const float* __restrict__ b, const __bf16* __restrict__ Wa1,
    const __bf16* __restrict__ Wd1, const float* __restrict__ Xa,
    const float* __restrict__ Xd, float* __restrict__ prevf,
    __bf16* __restrict__ prevb, __bf16* __restrict__ emab,
    unsigned int* __restrict__ bar) {
  __shared__ __bf16 lds_w[2 * 8 * 16 * DDIM];  // 256 KiB

  int wslot = threadIdx.x >> 5;
  int wid = blockIdx.x * 8 + wslot;  // 0..127
  int mt = wid >> 5;                 // 0..3  (64/16 batch tiles)
  int nt = wid & 31;                 // 0..31 (512/16 feature tiles)
  int lane = threadIdx.x & 31;
  int n = lane & 15, half = lane >> 4;
  int col = nt * 16 + n;
  int nt_base = (blockIdx.x * 8) & 31;  // first n-tile staged by this block

  {  // cooperative weight staging: rows [nt_base*16, nt_base*16+128) of both
    const int4* ga = (const int4*)(Wa1 + (size_t)(nt_base * 16) * DDIM);
    const int4* gd = (const int4*)(Wd1 + (size_t)(nt_base * 16) * DDIM);
    int4* lw = (int4*)lds_w;
    const int CH = (8 * 16 * DDIM) / 8;  // int4 per matrix = 8192
    for (int i = threadIdx.x; i < CH; i += 256) {
      lw[i]      = ga[i];
      lw[CH + i] = gd[i];
    }
  }
  __syncthreads();

  const __bf16* wa = lds_w + (size_t)((nt - nt_base) * 16) * DDIM;
  const __bf16* wd = wa + (size_t)(8 * 16) * DDIM;
  const __bf16* arow = prevb + (size_t)(mt * 16) * DDIM;
  unsigned int barcnt = 0;

  for (int t = 0; t < SEQ_T; ++t) {
    v8f aA = {}, aD = {};
    for (int k = 0; k < DDIM; k += 32) {
      v16bf af = load_frag_a(arow + k, DDIM);
      v16bf f1 = load_frag_b(wa + k, DDIM);
      v16bf f2 = load_frag_b(wd + k, DDIM);
      aA = wmma_bf16(af, f1, aA);
      aD = wmma_bf16(af, f2, aD);
    }
    float ev[8];
#pragma unroll
    for (int i = 0; i < 8; ++i) {
      int bi = mt * 16 + i + half * 8;
      int r = bi * SEQ_T + t;  // row in [32768, 512] tensors
      float al = tanh_f(aA[i] + Xa[r * DDIM + col]);
      float de = tanh_f(aD[i] + Xd[r * DDIM + col]);
      float bt = b[r * DDIM + col];
      float pv = prevf[bi * DDIM + col];
      float e = sigmoid_f(al * bt + (1.0f - al * de) * pv);
      ev[i] = e;
      emab[r * DDIM + col] = (__bf16)e;
    }
    barcnt += SCAN_BLOCKS;
    gsync(bar, barcnt);  // all reads of prev done
#pragma unroll
    for (int i = 0; i < 8; ++i) {
      int bi = mt * 16 + i + half * 8;
      prevf[bi * DDIM + col] = ev[i];
      prevb[bi * DDIM + col] = (__bf16)ev[i];
    }
    barcnt += SCAN_BLOCKS;
    gsync(bar, barcnt);  // all writes of prev done
  }
}

// ---------------- phase 3: out = silu(ema @ Wo^T + bo) -----------------------
// 2x2 register blocking: 4 WMMA per 4 fragment loads.
__global__ __launch_bounds__(256) void ema_proj_kernel(
    const __bf16* __restrict__ emab, const __bf16* __restrict__ Wob,
    const float* __restrict__ bo, float* __restrict__ out) {
  int wid = blockIdx.x * 8 + (threadIdx.x >> 5);
  int mtp = wid >> 4;  // 0..1023 (row-pair tiles)
  int ntp = wid & 15;  // 0..15   (col-pair tiles)
  int lane = threadIdx.x & 31;
  int n = lane & 15, half = lane >> 4;

  const __bf16* a0 = emab + (size_t)(mtp * 32) * DDIM;
  const __bf16* a1 = a0 + (size_t)16 * DDIM;
  const __bf16* w0 = Wob + (size_t)(ntp * 32) * DDIM;
  const __bf16* w1 = w0 + (size_t)16 * DDIM;
  v8f c00 = {}, c01 = {}, c10 = {}, c11 = {};
  for (int k = 0; k < DDIM; k += 32) {
    v16bf af0 = load_frag_a(a0 + k, DDIM);
    v16bf af1 = load_frag_a(a1 + k, DDIM);
    v16bf bf0 = load_frag_b(w0 + k, DDIM);
    v16bf bf1 = load_frag_b(w1 + k, DDIM);
    c00 = wmma_bf16(af0, bf0, c00);
    c01 = wmma_bf16(af0, bf1, c01);
    c10 = wmma_bf16(af1, bf0, c10);
    c11 = wmma_bf16(af1, bf1, c11);
  }
  int col0 = ntp * 32 + n, col1 = col0 + 16;
  float bo0 = bo[col0], bo1 = bo[col1];
#pragma unroll
  for (int i = 0; i < 8; ++i) {
    int row0 = mtp * 32 + i + half * 8;
    float x00 = c00[i] + bo0, x01 = c01[i] + bo1;
    float x10 = c10[i] + bo0, x11 = c11[i] + bo1;
    out[row0 * DDIM + col0] = x00 * sigmoid_f(x00);
    out[row0 * DDIM + col1] = x01 * sigmoid_f(x01);
    out[(row0 + 16) * DDIM + col0] = x10 * sigmoid_f(x10);
    out[(row0 + 16) * DDIM + col1] = x11 * sigmoid_f(x11);
  }
}

// ---------------- host launcher ----------------------------------------------
extern "C" void kernel_launch(void* const* d_in, const int* in_sizes, int n_in,
                              void* d_out, int out_size, void* d_ws, size_t ws_size,
                              hipStream_t stream) {
  const float* b  = (const float*)d_in[0];   // [64, 512, 512]
  const float* Wa = (const float*)d_in[1];   // [512, 1024]
  const float* ba = (const float*)d_in[2];   // [512]
  const float* Wd = (const float*)d_in[3];   // [512, 1024]
  const float* bd = (const float*)d_in[4];   // [512]
  const float* Wo = (const float*)d_in[5];   // [512, 512]
  const float* bo = (const float*)d_in[6];   // [512]
  float* out = (float*)d_out;                // [64, 512, 512]
  (void)in_sizes; (void)n_in; (void)out_size; (void)ws_size;

  uint8_t* base = (uint8_t*)d_ws;
  size_t off = 0;
  auto carve = [&](size_t bytes) -> void* {
    void* p = base + off;
    off = (off + bytes + 255) & ~(size_t)255;
    return p;
  };
  unsigned int* bar = (unsigned int*)carve(256);
  float*  prevf = (float*)carve((size_t)BATCH * DDIM * sizeof(float));
  __bf16* prevb = (__bf16*)carve((size_t)BATCH * DDIM * sizeof(__bf16));
  size_t zero_bytes = off;                 // bar + prev buffers start zeroed
  __bf16* Wa1b = (__bf16*)carve((size_t)DDIM * DDIM * 2);
  __bf16* Wa2b = (__bf16*)carve((size_t)DDIM * DDIM * 2);
  __bf16* Wd1b = (__bf16*)carve((size_t)DDIM * DDIM * 2);
  __bf16* Wd2b = (__bf16*)carve((size_t)DDIM * DDIM * 2);
  __bf16* Wob  = (__bf16*)carve((size_t)DDIM * DDIM * 2);
  __bf16* bb   = (__bf16*)carve((size_t)ROWS * DDIM * 2);
  float*  Xa   = (float*)carve((size_t)ROWS * DDIM * 4);
  float*  Xd   = (float*)carve((size_t)ROWS * DDIM * 4);
  __bf16* emab = (__bf16*)carve((size_t)ROWS * DDIM * 2);

  hipMemsetAsync(d_ws, 0, zero_bytes, stream);

  // conversions
  int nb = ROWS * DDIM;
  ema_cvt_kernel<<<nb / 256, 256, 0, stream>>>(b, bb, nb);
  int nw = DDIM * DDIM, wblk = (nw + 255) / 256;
  ema_cvt_slice_kernel<<<wblk, 256, 0, stream>>>(Wa, Wa1b, DDIM, DDIM, 2 * DDIM, 0);
  ema_cvt_slice_kernel<<<wblk, 256, 0, stream>>>(Wa, Wa2b, DDIM, DDIM, 2 * DDIM, DDIM);
  ema_cvt_slice_kernel<<<wblk, 256, 0, stream>>>(Wd, Wd1b, DDIM, DDIM, 2 * DDIM, 0);
  ema_cvt_slice_kernel<<<wblk, 256, 0, stream>>>(Wd, Wd2b, DDIM, DDIM, 2 * DDIM, DDIM);
  ema_cvt_slice_kernel<<<wblk, 256, 0, stream>>>(Wo, Wob, DDIM, DDIM, DDIM, 0);

  // phase 1: parallel precompute of input-dependent halves (2 m-tile blocking)
  int pre_blocks = (ROWS / 32) * (DDIM / 16) / 8;    // 4096
  ema_pre_kernel<<<pre_blocks, 256, 0, stream>>>(bb, Wa2b, Wd2b, ba, bd, Xa, Xd);

  // phase 2: persistent sequential scan (LDS-resident weights)
  ema_scan_kernel<<<SCAN_BLOCKS, 256, 0, stream>>>(b, Wa1b, Wd1b, Xa, Xd,
                                                   prevf, prevb, emab, bar);

  // phase 3: output projection + fused SiLU (2x2 blocking)
  int proj_blocks = (ROWS / 32) * (DDIM / 32) / 8;   // 2048
  ema_proj_kernel<<<proj_blocks, 256, 0, stream>>>(emab, Wob, bo, out);
}
